// FUsion_34497177322046
// MI455X (gfx1250) — matrix operand order repaired and checked by reading, not compiled, not run on hardware
//
#include <hip/hip_runtime.h>
#include <hip/hip_bf16.h>
#include <math.h>

// ---------------------------------------------------------------------------
// CDNA5 (gfx1250) VMamba fusion block.
//  - All dense projections: v_wmma_f32_16x16x32_f16, f32 accumulate.
//  - Weights pre-packed once per launch to f16 (N,K) rows -> B-fragment loads
//    are two 16B chunks per lane; row-major A fragments are float4 loads.
//  - Channel-major outputs stored as 2x float4 per lane (8 consecutive l).
//  - Selective scan: one wave32 lane per (b,k,d), 16-state in VGPRs, B/C via
//    double-buffered LDS (1 barrier/step), global_prefetch for future steps.
//  - Wave32 __shfl_xor reductions for LayerNorm.
// Workspace: ~126.3 MB float arena in d_ws.
// ---------------------------------------------------------------------------

typedef __attribute__((ext_vector_type(16))) _Float16 v16h;
typedef __attribute__((ext_vector_type(8)))  _Float16 v8h_t;
typedef __attribute__((ext_vector_type(8)))  float    v8f;

#define DEV __device__ __forceinline__

constexpr float EPSV = 1e-5f;
constexpr int   Bx = 4;

// ---- workspace layout (float offsets) -------------------------------------
constexpr size_t OFF_XI    = 0;          // (B,DI,L)
constexpr size_t OFF_Z     = 2097152;    // (B*L,DI)
constexpr size_t OFF_XC    = 4194304;    // (B,DI,L)
constexpr size_t OFF_XDBL  = 6291456;    // (B,K,48,L)
constexpr size_t OFF_DELTA = 9437184;    // (B,K,DI,L)
constexpr size_t OFF_OY    = 17825792;   // (B,K,DI,L)
constexpr size_t OFF_YC    = 26214400;   // (B*L,DI)
constexpr size_t POOL_END  = 28311552;
constexpr size_t OFF_Y1    = POOL_END;
constexpr size_t OFF_Y2    = POOL_END + 1048576;
constexpr size_t OFF_MEG   = POOL_END + 2097152;
// attention phase (reuses pool)
constexpr size_t OFF_T1RAW = 0;
constexpr size_t OFF_T1LN  = 1048576;
constexpr size_t OFF_T2LN  = 2097152;
constexpr size_t OFF_Q     = 3145728;
constexpr size_t OFF_KV    = 4194304;
constexpr size_t OFF_O     = 6291456;
constexpr size_t OFF_TT    = 7340032;
constexpr size_t OFF_TLN   = 8388608;
constexpr size_t OFF_HID   = 9437184;
constexpr size_t OFF_TOUT  = 13631488;
// pinwheel phase (reuses pool)
constexpr size_t OFF_RES   = 0;
constexpr size_t OFF_YCAT  = 1048576;    // 4*64*65*65
constexpr size_t OFF_PWOUT = 2162688;
constexpr size_t OFF_BNT   = 3211264;
// packed f16 weights (half offsets, base = ws + OFF_PACK floats)
constexpr size_t OFF_PACK  = 31457280;   // floats
constexpr size_t PH_INPROJ_R = 0;        // (256,64)
constexpr size_t PH_INPROJ_D = 16384;
constexpr size_t PH_XPW_R    = 32768;    // (4,48,128)
constexpr size_t PH_XPW_D    = 57344;
constexpr size_t PH_OUTP_R   = 81920;    // (64,128)
constexpr size_t PH_OUTP_D   = 90112;
constexpr size_t PH_QW       = 98304;    // (64,64)
constexpr size_t PH_KVW      = 102400;   // (128,64)
constexpr size_t PH_PROJW    = 110592;   // (64,64)
constexpr size_t PH_FC1      = 114688;   // (256,64)
constexpr size_t PH_FC2      = 131072;   // (64,256)
constexpr size_t PH_CATW     = 147456;   // (64,256)
constexpr size_t PH_RESW     = 163840;   // (64,64)
constexpr size_t PH_BNW      = 167936;   // (64,64)

// ---- helpers ---------------------------------------------------------------
DEV float wred32(float x) {
#pragma unroll
  for (int m = 16; m > 0; m >>= 1) x += __shfl_xor(x, m, 32);
  return x;
}
DEV float siluf(float x) { return x / (1.0f + __expf(-x)); }

DEV int srcl(int k, int l) {
  int lt = ((l & 63) << 6) | (l >> 6);
  switch (k & 3) {
    case 0:  return l;
    case 1:  return lt;
    case 2:  return 4095 - l;
    default: { int lf = 4095 - l; return ((lf & 63) << 6) | (lf >> 6); }
  }
}

// ---- WMMA fragment loaders (CDNA5 16x16x32 f16 VGPR layouts) ---------------
DEV v16h load_a_rm(const float* __restrict__ A, int m0, int Kd, int kk) {
  int lane = threadIdx.x & 31, row = lane & 15, kh = lane >> 4;
  const float* ap = A + (size_t)(m0 + row) * Kd + kk + kh * 8;
  float4 x0 = *(const float4*)(ap);
  float4 x1 = *(const float4*)(ap + 4);
  float4 x2 = *(const float4*)(ap + 16);
  float4 x3 = *(const float4*)(ap + 20);
  v16h a;
  a[0]  = (_Float16)x0.x; a[1]  = (_Float16)x0.y; a[2]  = (_Float16)x0.z; a[3]  = (_Float16)x0.w;
  a[4]  = (_Float16)x1.x; a[5]  = (_Float16)x1.y; a[6]  = (_Float16)x1.z; a[7]  = (_Float16)x1.w;
  a[8]  = (_Float16)x2.x; a[9]  = (_Float16)x2.y; a[10] = (_Float16)x2.z; a[11] = (_Float16)x2.w;
  a[12] = (_Float16)x3.x; a[13] = (_Float16)x3.y; a[14] = (_Float16)x3.z; a[15] = (_Float16)x3.w;
  return a;
}
template <typename F> DEV v16h load_a_gather(F f) {
  int lane = threadIdx.x & 31, kh = lane >> 4;
  v16h a;
#pragma unroll
  for (int j = 0; j < 8; ++j) {
    int k0 = ((j & 4) ? 16 : 0) + kh * 8 + 2 * (j & 3);
    a[2 * j]     = (_Float16)f(k0);
    a[2 * j + 1] = (_Float16)f(k0 + 1);
  }
  return a;
}
DEV v16h load_b_packed(const _Float16* __restrict__ Wp, int n0, int Kd, int kk) {
  int lane = threadIdx.x & 31, col = lane & 15, kh = lane >> 4;
  const _Float16* p = Wp + (size_t)(n0 + col) * Kd + kk + kh * 8;
  v8h_t lo = *(const v8h_t*)(p);
  v8h_t hi = *(const v8h_t*)(p + 16);
  return __builtin_shufflevector(lo, hi, 0, 1, 2, 3, 4, 5, 6, 7,
                                 8, 9, 10, 11, 12, 13, 14, 15);
}
// channel-major store: lane's 8 acc values are 8 consecutive l -> 2x float4
DEV void store_cm8(float* __restrict__ po, const float* v) {
  float4 s0, s1;
  s0.x = v[0]; s0.y = v[1]; s0.z = v[2]; s0.w = v[3];
  s1.x = v[4]; s1.y = v[5]; s1.z = v[6]; s1.w = v[7];
  *(float4*)(po)     = s0;
  *(float4*)(po + 4) = s1;
}

// ---- weight pack: f32 (Kd,N) or (N,Kd) -> f16 (N,Kd), zero-pad rows >=Nsrc --
__global__ void k_packw(const float* __restrict__ in, _Float16* __restrict__ out,
                        int N, int Kd, int Nsrc, int trans) {
  int idx = blockIdx.x * 256 + threadIdx.x;
  if (idx >= N * Kd) return;
  int n = idx / Kd, k = idx - n * Kd;
  float v = 0.0f;
  if (n < Nsrc) v = trans ? in[(size_t)k * Nsrc + n] : in[(size_t)n * Kd + k];
  out[idx] = (_Float16)v;
}

// ---------------------------------------------------------------------------
// Templated GEMM: out = act(A @ Wp^T + bias) + resid     (Wp packed (N,Kd) f16)
//  AMODE: 0 -> A[m*Kd+k];  1 -> A[(b*Kd+k)*4096 + l], m=b*4096+l
//  OMODE: 0 -> out[m*N+n]; 1 -> out[(b*N+n)*4096+l]  (2x float4 per lane)
//  ACT  : 0 none; 1 tanh-gelu
// ---------------------------------------------------------------------------
template <int AMODE, int ACT, int OMODE>
__global__ void k_gemm_t(const float* __restrict__ A, const _Float16* __restrict__ Wp,
                         const float* __restrict__ bias, const float* __restrict__ resid,
                         float* __restrict__ out, int M, int N, int Kd) {
  int tile = blockIdx.x * (blockDim.x >> 5) + (threadIdx.x >> 5);
  int Nt = N >> 4;
  if (tile >= (M >> 4) * Nt) return;
  int mt = tile / Nt, nt = tile - mt * Nt;
  int m0 = mt << 4, n0 = nt << 4;
  int lane = threadIdx.x & 31, row = lane & 15, kh = lane >> 4;
  v8f acc = {};
  if (AMODE == 0) {
    for (int kk = 0; kk < Kd; kk += 32) {
      v16h af = load_a_rm(A, m0, Kd, kk);
      v16h bf = load_b_packed(Wp, n0, Kd, kk);
      acc = __builtin_amdgcn_wmma_f32_16x16x32_f16(false, af, false, bf,
                                                   (short)0, acc, false, false);
    }
  } else {
    int m = m0 + row, b = m >> 12, l = m & 4095;
    const float* ap = A + (((size_t)b * Kd) << 12) + l;
    for (int kk = 0; kk < Kd; kk += 32) {
      v16h af = load_a_gather([&](int kq) -> float {
        return ap[(size_t)(kk + kq) << 12];
      });
      v16h bf = load_b_packed(Wp, n0, Kd, kk);
      acc = __builtin_amdgcn_wmma_f32_16x16x32_f16(false, af, false, bf,
                                                   (short)0, acc, false, false);
    }
  }
  // epilogue: acc[r] corresponds to m = m0 + 8*kh + r, n = n0 + col
  int col = lane & 15, n = n0 + col;
  int mbase = m0 + (kh << 3);
  float v[8];
  float bn = bias ? bias[n] : 0.0f;
#pragma unroll
  for (int r = 0; r < 8; ++r) {
    float x = acc[r] + bn;
    if (ACT == 1)
      x = 0.5f * x * (1.0f + tanhf(0.7978845608f * (x + 0.044715f * x * x * x)));
    v[r] = x;
  }
  if (OMODE == 0) {
    float* po = out + (size_t)mbase * N + n;
    if (resid) {
      const float* pr = resid + (size_t)mbase * N + n;
#pragma unroll
      for (int r = 0; r < 8; ++r) po[(size_t)r * N] = v[r] + pr[(size_t)r * N];
    } else {
#pragma unroll
      for (int r = 0; r < 8; ++r) po[(size_t)r * N] = v[r];
    }
  } else {
    int b = mbase >> 12, l0 = mbase & 4095;
    store_cm8(out + (((size_t)(b * N + n)) << 12) + l0, v);
  }
}

// xz = x_perm @ in_proj  (A: x in (B,64,L) channel-major; split xi/z outputs)
__global__ void k_gemm_xz(const float* __restrict__ X, const _Float16* __restrict__ Wp,
                          float* __restrict__ XI, float* __restrict__ Z) {
  int tile = blockIdx.x * 8 + (threadIdx.x >> 5);   // Mt=1024, Nt=16
  if (tile >= 1024 * 16) return;
  int mt = tile >> 4, nt = tile & 15;
  int m0 = mt << 4, n0 = nt << 4;
  int lane = threadIdx.x & 31, row = lane & 15, kh = lane >> 4;
  int m = m0 + row, b = m >> 12, l = m & 4095;
  const float* ap = X + (((size_t)b * 64) << 12) + l;
  v8f acc = {};
#pragma unroll
  for (int kk = 0; kk < 64; kk += 32) {
    v16h af = load_a_gather([&](int kq) -> float {
      return ap[(size_t)(kk + kq) << 12];
    });
    v16h bf = load_b_packed(Wp, n0, 64, kk);
    acc = __builtin_amdgcn_wmma_f32_16x16x32_f16(false, af, false, bf,
                                                 (short)0, acc, false, false);
  }
  int col = lane & 15, n = n0 + col;
  int mbase = m0 + (kh << 3);
  int bb = mbase >> 12, l0 = mbase & 4095;
  float v[8];
#pragma unroll
  for (int r = 0; r < 8; ++r) v[r] = acc[r];
  if (n0 < 128) {                       // wave-uniform: xi half (chan-major)
    store_cm8(XI + (((size_t)(bb * 128 + n)) << 12) + l0, v);
  } else {                              // z half (row-major, 128 cols)
    float* po = Z + (size_t)mbase * 128 + (n - 128);
#pragma unroll
    for (int r = 0; r < 8; ++r) po[(size_t)r * 128] = v[r];
  }
}

// x_dbl = xs @ x_proj_w^T  (M = B*K*L; N padded to 48; A gathered by dir map)
__global__ void k_gemm_xdbl(const float* __restrict__ XC, const _Float16* __restrict__ XPW,
                            float* __restrict__ XDBL) {
  int tile = blockIdx.x * 8 + (threadIdx.x >> 5);   // Mt=4096, Nt=3
  if (tile >= 4096 * 3) return;
  int mt = tile / 3, nt = tile - mt * 3;
  int m0 = mt << 4, n0 = nt << 4;
  int bk = m0 >> 12, b = bk >> 2, k = bk & 3;       // uniform per tile
  const _Float16* wk = XPW + (size_t)k * 48 * 128;
  int lane = threadIdx.x & 31, row = lane & 15, kh = lane >> 4;
  int l = (m0 + row) & 4095;
  const float* ap = XC + (((size_t)b * 128) << 12) + srcl(k, l);
  v8f acc = {};
#pragma unroll
  for (int kk = 0; kk < 128; kk += 32) {
    v16h af = load_a_gather([&](int kq) -> float {
      return ap[(size_t)(kk + kq) << 12];
    });
    v16h bf = load_b_packed(wk, n0, 128, kk);
    acc = __builtin_amdgcn_wmma_f32_16x16x32_f16(false, af, false, bf,
                                                 (short)0, acc, false, false);
  }
  int col = lane & 15, n = n0 + col;
  int l0 = (m0 + (kh << 3)) & 4095;
  float v[8];
#pragma unroll
  for (int r = 0; r < 8; ++r) v[r] = acc[r];
  store_cm8(XDBL + (((size_t)(bk * 48 + n)) << 12) + l0, v);
}

// 2x2 "cat" conv as im2col WMMA GEMM (Kd = 64ch*4 taps) + eval-BN + SiLU
__global__ void k_catconv(const float* __restrict__ YCAT, const _Float16* __restrict__ Wp,
                          const float* __restrict__ g, const float* __restrict__ bb,
                          float* __restrict__ PW) {
  int tile = blockIdx.x * 8 + (threadIdx.x >> 5);   // Mt=1024, Nt=4
  if (tile >= 1024 * 4) return;
  int mt = tile >> 2, nt = tile & 3;
  int m0 = mt << 4, n0 = nt << 4;
  int lane = threadIdx.x & 31, row = lane & 15, kh = lane >> 4;
  int m = m0 + row, b = m >> 12, l = m & 4095;
  int h = l >> 6, w = l & 63;
  const float* base = YCAT + (size_t)(b * 64) * 4225 + h * 65 + w;
  v8f acc = {};
#pragma unroll
  for (int kk = 0; kk < 256; kk += 32) {
    v16h af = load_a_gather([&](int kq) -> float {
      int kq2 = kk + kq, c = kq2 >> 2, t = kq2 & 3;
      return base[(size_t)c * 4225 + (t >> 1) * 65 + (t & 1)];
    });
    v16h bf = load_b_packed(Wp, n0, 256, kk);
    acc = __builtin_amdgcn_wmma_f32_16x16x32_f16(false, af, false, bf,
                                                 (short)0, acc, false, false);
  }
  int col = lane & 15, n = n0 + col;
  int mbase = m0 + (kh << 3);
  int bb2 = mbase >> 12, l0 = mbase & 4095;
  float rs = rsqrtf(1.0f + EPSV);
  float sg = g[n] * rs, sb = bb[n];
  float v[8];
#pragma unroll
  for (int r = 0; r < 8; ++r) v[r] = siluf(acc[r] * sg + sb);
  store_cm8(PW + (((size_t)(bb2 * 64 + n)) << 12) + l0, v);
}

// depthwise 3x3 SAME + bias + SiLU
__global__ void k_dwconv(const float* __restrict__ XI, const float* __restrict__ cw,
                         const float* __restrict__ cb, float* __restrict__ XC) {
  int idx = blockIdx.x * 256 + threadIdx.x;
  int l = idx & 4095, bd = idx >> 12, d = bd & 127;
  int h = l >> 6, w = l & 63;
  const float* xin = XI + ((size_t)bd << 12);
  const float* wp = cw + d * 9;
  float acc = cb[d];
#pragma unroll
  for (int ky = 0; ky < 3; ++ky) {
    int ih = h + ky - 1;
    if ((unsigned)ih > 63u) continue;
#pragma unroll
    for (int kx = 0; kx < 3; ++kx) {
      int iw = w + kx - 1;
      if ((unsigned)iw > 63u) continue;
      acc += xin[ih * 64 + iw] * wp[ky * 3 + kx];
    }
  }
  XC[idx] = siluf(acc);
}

// delta = softplus(dts @ dt_proj_w^T + dt_proj_b)
__global__ void k_delta(const float* __restrict__ XDBL, const float* __restrict__ dtw,
                        const float* __restrict__ dtb, float* __restrict__ DELTA) {
  size_t idx = (size_t)blockIdx.x * 256 + threadIdx.x;
  int l = (int)(idx & 4095);
  int kd = (int)(idx >> 12), d = kd & 127, bk = kd >> 7, k = bk & 3;
  float s = dtb[k * 128 + d];
#pragma unroll
  for (int r = 0; r < 4; ++r)
    s += XDBL[((size_t)(bk * 48 + r) << 12) + l] * dtw[(size_t)(k * 128 + d) * 4 + r];
  DELTA[idx] = (s > 20.0f) ? s : log1pf(__expf(s));
}

// selective scan: one lane per (b,k,d); 16-wide state in VGPRs;
// B/C double-buffered through LDS -> single barrier per step.
__global__ void k_scan(const float* __restrict__ XC, const float* __restrict__ DELTA,
                       const float* __restrict__ XDBL, const float* __restrict__ Alog,
                       const float* __restrict__ Dp, float* __restrict__ OY) {
  __shared__ float sBC[2][32];
  int bk = blockIdx.x, b = bk >> 2, k = bk & 3;
  int d = threadIdx.x;
  float a[16], h[16];
#pragma unroll
  for (int n = 0; n < 16; ++n) {
    a[n] = -__expf(Alog[((size_t)(k * 128 + d) << 4) + n]);
    h[n] = 0.0f;
  }
  float Dv = Dp[k * 128 + d];
  const float* up = XC + ((size_t)(b * 128 + d) << 12);
  const float* dp = DELTA + ((size_t)(bk * 128 + d) << 12);
  const float* bc = XDBL + ((size_t)(bk * 48 + 4) << 12);
  float* op = OY + ((size_t)(bk * 128 + d) << 12);
  if (threadIdx.x < 32) sBC[0][threadIdx.x] = bc[((size_t)threadIdx.x << 12)];
  __syncthreads();
  for (int l = 0; l < 4096; ++l) {
    int cur = l & 1;
    if (threadIdx.x < 32 && l + 1 < 4096) {
      sBC[cur ^ 1][threadIdx.x] = bc[((size_t)threadIdx.x << 12) + l + 1];
      if (l + 16 < 4096)
        __builtin_prefetch(&bc[((size_t)threadIdx.x << 12) + l + 16], 0, 1);
    }
    float u = up[srcl(k, l)];
    float dt = dp[l];
    float du = dt * u;
    float y = 0.0f;
#pragma unroll
    for (int n = 0; n < 16; ++n) {
      h[n] = h[n] * __expf(dt * a[n]) + du * sBC[cur][n];
      y += h[n] * sBC[cur][16 + n];
    }
    op[l] = y + u * Dv;
    __syncthreads();
  }
}

// combine 4 scan directions, LN over DI, gate with SiLU(z)
__global__ void k_combine(const float* __restrict__ OY, const float* __restrict__ Z,
                          const float* __restrict__ lng, const float* __restrict__ lnb,
                          float* __restrict__ YC) {
  int lane = threadIdx.x & 31;
  int tok = blockIdx.x * 8 + (threadIdx.x >> 5);
  int b = tok >> 12, l = tok & 4095;
  int lt = ((l & 63) << 6) | (l >> 6);
  float val[4], s = 0.0f;
#pragma unroll
  for (int i = 0; i < 4; ++i) {
    int d = lane + 32 * i;
    const float* o0 = OY + ((size_t)((b * 4 + 0) * 128 + d) << 12);
    const float* o1 = OY + ((size_t)((b * 4 + 1) * 128 + d) << 12);
    const float* o2 = OY + ((size_t)((b * 4 + 2) * 128 + d) << 12);
    const float* o3 = OY + ((size_t)((b * 4 + 3) * 128 + d) << 12);
    float v = o0[l] + o2[4095 - l] + o1[lt] + o3[4095 - lt];
    val[i] = v;
    s += v;
  }
  float mean = wred32(s) * (1.0f / 128.0f);
  float vs = 0.0f;
#pragma unroll
  for (int i = 0; i < 4; ++i) { val[i] -= mean; vs += val[i] * val[i]; }
  float rstd = rsqrtf(wred32(vs) * (1.0f / 128.0f) + EPSV);
#pragma unroll
  for (int i = 0; i < 4; ++i) {
    int d = lane + 32 * i;
    float z = Z[(size_t)tok * 128 + d];
    YC[(size_t)tok * 128 + d] = (val[i] * rstd * lng[d] + lnb[d]) * siluf(z);
  }
}

// tokenize + two LayerNorms; keep raw t1 for residual
__global__ void k_tok_ln(const float* __restrict__ Y1, const float* __restrict__ Y2,
                         const float* n1g, const float* n1b,
                         const float* n2g, const float* n2b,
                         float* __restrict__ T1RAW, float* __restrict__ T1LN,
                         float* __restrict__ T2LN) {
  int lane = threadIdx.x & 31;
  int t = blockIdx.x * 8 + (threadIdx.x >> 5);
  int g = t >> 6, n = t & 63;
  int b = g >> 6, i = (g >> 3) & 7, j = g & 7;
  int l = ((n >> 3) * 8 + i) * 64 + ((n & 7) * 8 + j);
  int c0 = lane, c1 = lane + 32;
  float a0 = Y1[((size_t)(b * 64 + c0) << 12) + l];
  float a1 = Y1[((size_t)(b * 64 + c1) << 12) + l];
  float e0 = Y2[((size_t)(b * 64 + c0) << 12) + l];
  float e1 = Y2[((size_t)(b * 64 + c1) << 12) + l];
  T1RAW[(size_t)t * 64 + c0] = a0;
  T1RAW[(size_t)t * 64 + c1] = a1;
  float m1 = wred32(a0 + a1) * (1.0f / 64.0f);
  float d0 = a0 - m1, d1 = a1 - m1;
  float r1 = rsqrtf(wred32(d0 * d0 + d1 * d1) * (1.0f / 64.0f) + EPSV);
  T1LN[(size_t)t * 64 + c0] = d0 * r1 * n1g[c0] + n1b[c0];
  T1LN[(size_t)t * 64 + c1] = d1 * r1 * n1g[c1] + n1b[c1];
  float m2 = wred32(e0 + e1) * (1.0f / 64.0f);
  float f0 = e0 - m2, f1 = e1 - m2;
  float r2 = rsqrtf(wred32(f0 * f0 + f1 * f1) * (1.0f / 64.0f) + EPSV);
  T2LN[(size_t)t * 64 + c0] = f0 * r2 * n2g[c0] + n2b[c0];
  T2LN[(size_t)t * 64 + c1] = f1 * r2 * n2g[c1] + n2b[c1];
}

// LN over 64 (n3)
__global__ void k_ln64(const float* __restrict__ X, const float* g,
                       const float* bb, float* __restrict__ Y) {
  int lane = threadIdx.x & 31;
  int t = blockIdx.x * 8 + (threadIdx.x >> 5);
  float a0 = X[(size_t)t * 64 + lane], a1 = X[(size_t)t * 64 + lane + 32];
  float m = wred32(a0 + a1) * (1.0f / 64.0f);
  float d0 = a0 - m, d1 = a1 - m;
  float r = rsqrtf(wred32(d0 * d0 + d1 * d1) * (1.0f / 64.0f) + EPSV);
  Y[(size_t)t * 64 + lane]      = d0 * r * g[lane] + bb[lane];
  Y[(size_t)t * 64 + lane + 32] = d1 * r * g[lane + 32] + bb[lane + 32];
}

// 8-head attention, hd=8, 64 tokens/group: one wave per (group, head)
__global__ void k_attn(const float* __restrict__ Q, const float* __restrict__ KV,
                       float* __restrict__ O) {
  __shared__ float sk[512], sv[512], ss[4096];
  int gh = blockIdx.x, g = gh >> 3, h = gh & 7;
  int lane = threadIdx.x;
  for (int i = lane; i < 512; i += 32) {
    int m = i >> 3, d = i & 7;
    sk[i] = KV[(size_t)(g * 64 + m) * 128 + h * 8 + d];
    sv[i] = KV[(size_t)(g * 64 + m) * 128 + 64 + h * 8 + d];
  }
  __syncthreads();
  const float scale = 0.35355339059f;
#pragma unroll
  for (int rr = 0; rr < 2; ++rr) {
    int r = lane + rr * 32;
    float q[8];
#pragma unroll
    for (int d = 0; d < 8; ++d) q[d] = Q[(size_t)(g * 64 + r) * 64 + h * 8 + d];
    float mx = -1e30f;
    for (int m = 0; m < 64; ++m) {
      float s = 0.0f;
#pragma unroll
      for (int d = 0; d < 8; ++d) s += q[d] * sk[m * 8 + d];
      s *= scale;
      ss[r * 64 + m] = s;
      mx = fmaxf(mx, s);
    }
    float sum = 0.0f;
    for (int m = 0; m < 64; ++m) {
      float e = __expf(ss[r * 64 + m] - mx);
      ss[r * 64 + m] = e;
      sum += e;
    }
    float inv = 1.0f / sum;
    float acc[8] = {};
    for (int m = 0; m < 64; ++m) {
      float p = ss[r * 64 + m] * inv;
#pragma unroll
      for (int d = 0; d < 8; ++d) acc[d] += p * sv[m * 8 + d];
    }
#pragma unroll
    for (int d = 0; d < 8; ++d) O[(size_t)(g * 64 + r) * 64 + h * 8 + d] = acc[d];
  }
}

// un-tokenize (g,n,c) -> (b,c,h,w)
__global__ void k_untok(const float* __restrict__ T, float* __restrict__ MEG) {
  int idx = blockIdx.x * 256 + threadIdx.x;
  int c = idx & 63, t = idx >> 6;
  int g = t >> 6, n = t & 63;
  int b = g >> 6, i = (g >> 3) & 7, j = g & 7;
  int l = ((n >> 3) * 8 + i) * 64 + ((n & 7) * 8 + j);
  MEG[((size_t)(b * 64 + c) << 12) + l] = T[idx];
}

// pinwheel: 4 asymmetric-pad branches into (B,64,65,65), eval-BN + SiLU
__global__ void k_pinwheel(const float* __restrict__ X,
                           const float* cw, const float* cwg, const float* cwb,
                           const float* ch, const float* chg, const float* chb,
                           float* __restrict__ Y) {
  int idx = blockIdx.x * 256 + threadIdx.x;
  if (idx >= Bx * 64 * 65 * 65) return;
  int ow = idx % 65, t = idx / 65;
  int oh = t % 65;  t /= 65;
  int oc = t & 63;
  int br = oc >> 4, oq = oc & 15;
  const float* wp = (br < 2 ? cw : ch) + (size_t)oq * 64 * 3;
  const float* xin = X + ((size_t)(t >> 6) << 18);
  float acc = 0.0f;
  for (int ic = 0; ic < 64; ++ic) {
    const float* xi = xin + ((size_t)ic << 12);
#pragma unroll
    for (int kt = 0; kt < 3; ++kt) {
      int ih, iw;
      if      (br == 0) { ih = oh - 1;      iw = ow - 3 + kt; }
      else if (br == 1) { ih = oh;          iw = ow + kt;     }
      else if (br == 2) { ih = oh - 3 + kt; iw = ow;          }
      else              { ih = oh + kt;     iw = ow - 1;      }
      float xv = ((unsigned)ih < 64u && (unsigned)iw < 64u) ? xi[ih * 64 + iw] : 0.0f;
      acc += xv * wp[ic * 3 + kt];
    }
  }
  float rs = rsqrtf(1.0f + EPSV);
  float g = (br < 2 ? cwg : chg)[oq], bb = (br < 2 ? cwb : chb)[oq];
  Y[idx] = siluf(acc * (g * rs) + bb);
}

// out = bn(relu(bn(bnconv)) + residual, norm)
__global__ void k_final(const float* __restrict__ BNT, const float* __restrict__ RES,
                        const float* bg, const float* bb,
                        const float* ng, const float* nb, float* __restrict__ out) {
  int idx = blockIdx.x * 256 + threadIdx.x;
  int o = (idx >> 12) & 63;
  float rs = rsqrtf(1.0f + EPSV);
  float v = fmaxf(0.0f, BNT[idx] * (bg[o] * rs) + bb[o]);
  out[idx] = (v + RES[idx]) * (ng[o] * rs) + nb[o];
}

// ---------------------------------------------------------------------------
static void launch_ss2d(const float* x, void* const* din, int pb, float* ws,
                        const _Float16* inprojP, const _Float16* xpwP,
                        const _Float16* outpP, float* Yout, hipStream_t st) {
  const float* conv_w = (const float*)din[pb + 1];
  const float* conv_b = (const float*)din[pb + 2];
  const float* dtw    = (const float*)din[pb + 4];
  const float* dtb    = (const float*)din[pb + 5];
  const float* Alog   = (const float*)din[pb + 6];
  const float* Dp     = (const float*)din[pb + 7];
  const float* lng    = (const float*)din[pb + 8];
  const float* lnb    = (const float*)din[pb + 9];
  float *XI = ws + OFF_XI, *Z = ws + OFF_Z, *XC = ws + OFF_XC;
  float *XDBL = ws + OFF_XDBL, *DELTA = ws + OFF_DELTA;
  float *OY = ws + OFF_OY, *YC = ws + OFF_YC;

  k_gemm_xz  <<<2048, 256, 0, st>>>(x, inprojP, XI, Z);
  k_dwconv   <<<8192, 256, 0, st>>>(XI, conv_w, conv_b, XC);
  k_gemm_xdbl<<<1536, 256, 0, st>>>(XC, xpwP, XDBL);
  k_delta    <<<32768, 256, 0, st>>>(XDBL, dtw, dtb, DELTA);
  k_scan     <<<16, 128, 0, st>>>(XC, DELTA, XDBL, Alog, Dp, OY);
  k_combine  <<<2048, 256, 0, st>>>(OY, Z, lng, lnb, YC);
  k_gemm_t<0, 0, 1><<<512, 256, 0, st>>>(YC, outpP, nullptr, nullptr, Yout,
                                         16384, 64, 128);
}

extern "C" void kernel_launch(void* const* d_in, const int* in_sizes, int n_in,
                              void* d_out, int out_size, void* d_ws, size_t ws_size,
                              hipStream_t stream) {
  (void)in_sizes; (void)n_in; (void)out_size; (void)ws_size;
  float* ws = (float*)d_ws;
  _Float16* HB = (_Float16*)(ws + OFF_PACK);

  // ---- pack all GEMM weights to f16 (N,Kd) rows ---------------------------
  auto P = [&](int idx) { return (const float*)d_in[idx]; };
  k_packw<<<64, 256, 0, stream>>>(P(2),  HB + PH_INPROJ_R, 256, 64, 256, 1);
  k_packw<<<64, 256, 0, stream>>>(P(13), HB + PH_INPROJ_D, 256, 64, 256, 1);
  for (int k = 0; k < 4; ++k) {
    k_packw<<<24, 256, 0, stream>>>(P(2 + 3)  + (size_t)k * 36 * 128,
                                    HB + PH_XPW_R + (size_t)k * 48 * 128, 48, 128, 36, 0);
    k_packw<<<24, 256, 0, stream>>>(P(13 + 3) + (size_t)k * 36 * 128,
                                    HB + PH_XPW_D + (size_t)k * 48 * 128, 48, 128, 36, 0);
  }
  k_packw<<<32, 256, 0, stream>>>(P(12), HB + PH_OUTP_R, 64, 128, 64, 1);
  k_packw<<<32, 256, 0, stream>>>(P(23), HB + PH_OUTP_D, 64, 128, 64, 1);
  k_packw<<<16, 256, 0, stream>>>(P(28), HB + PH_QW,    64, 64, 64, 1);
  k_packw<<<32, 256, 0, stream>>>(P(30), HB + PH_KVW,  128, 64, 128, 1);
  k_packw<<<16, 256, 0, stream>>>(P(32), HB + PH_PROJW, 64, 64, 64, 1);
  k_packw<<<64, 256, 0, stream>>>(P(36), HB + PH_FC1,  256, 64, 256, 1);
  k_packw<<<64, 256, 0, stream>>>(P(38), HB + PH_FC2,   64, 256, 64, 1);
  k_packw<<<64, 256, 0, stream>>>(P(46), HB + PH_CATW,  64, 256, 64, 0);
  k_packw<<<16, 256, 0, stream>>>(P(49), HB + PH_RESW,  64, 64, 64, 0);
  k_packw<<<16, 256, 0, stream>>>(P(50), HB + PH_BNW,   64, 64, 64, 0);

  // ---- two SS2D branches --------------------------------------------------
  launch_ss2d((const float*)d_in[0], d_in, 2, ws, HB + PH_INPROJ_R,
              HB + PH_XPW_R, HB + PH_OUTP_R, ws + OFF_Y1, stream);
  launch_ss2d((const float*)d_in[1], d_in, 13, ws, HB + PH_INPROJ_D,
              HB + PH_XPW_D, HB + PH_OUTP_D, ws + OFF_Y2, stream);

  // ---- sparse cross attention --------------------------------------------
  float *T1RAW = ws + OFF_T1RAW, *T1LN = ws + OFF_T1LN, *T2LN = ws + OFF_T2LN;
  float *Q = ws + OFF_Q, *KV = ws + OFF_KV, *O = ws + OFF_O;
  float *TT = ws + OFF_TT, *TLN = ws + OFF_TLN, *HID = ws + OFF_HID;
  float *TOUT = ws + OFF_TOUT, *MEG = ws + OFF_MEG;

  k_tok_ln<<<2048, 256, 0, stream>>>(ws + OFF_Y1, ws + OFF_Y2,
                                     P(24), P(25), P(26), P(27), T1RAW, T1LN, T2LN);
  k_gemm_t<0, 0, 0><<<512, 256, 0, stream>>>(T1LN, HB + PH_QW, P(29), nullptr, Q,
                                             16384, 64, 64);
  k_gemm_t<0, 0, 0><<<1024, 256, 0, stream>>>(T2LN, HB + PH_KVW, P(31), nullptr, KV,
                                              16384, 128, 64);
  k_attn<<<2048, 32, 0, stream>>>(Q, KV, O);
  k_gemm_t<0, 0, 0><<<512, 256, 0, stream>>>(O, HB + PH_PROJW, P(33), T1RAW, TT,
                                             16384, 64, 64);
  k_ln64<<<2048, 256, 0, stream>>>(TT, P(34), P(35), TLN);
  k_gemm_t<0, 1, 0><<<2048, 256, 0, stream>>>(TLN, HB + PH_FC1, P(37), nullptr, HID,
                                              16384, 256, 64);
  k_gemm_t<0, 0, 0><<<512, 256, 0, stream>>>(HID, HB + PH_FC2, P(39), TT, TOUT,
                                             16384, 64, 256);
  k_untok<<<4096, 256, 0, stream>>>(TOUT, MEG);

  // ---- pinwheel + residual + norms ---------------------------------------
  float *RES = ws + OFF_RES, *YCAT = ws + OFF_YCAT;
  float *PW = ws + OFF_PWOUT, *BNT = ws + OFF_BNT;

  k_gemm_t<1, 0, 1><<<512, 256, 0, stream>>>(MEG, HB + PH_RESW, nullptr, nullptr,
                                             RES, 16384, 64, 64);
  k_pinwheel<<<4225, 256, 0, stream>>>(MEG, P(40), P(41), P(42), P(43), P(44), P(45),
                                       YCAT);
  k_catconv<<<512, 256, 0, stream>>>(YCAT, HB + PH_CATW, P(47), P(48), PW);
  k_gemm_t<1, 0, 1><<<512, 256, 0, stream>>>(PW, HB + PH_BNW, nullptr, nullptr,
                                             BNT, 16384, 64, 64);
  k_final<<<4096, 256, 0, stream>>>(BNT, RES, P(51), P(52), P(53), P(54),
                                    (float*)d_out);
}
